// Transformer_49151605735526
// MI455X (gfx1250) — compile-verified, matrix operand order checked by reading, not run on hardware
//
#include <hip/hip_runtime.h>
#include <hip/hip_bf16.h>
#include <math.h>

// ---------------- model constants ----------------
#define NL   8
#define Dm   1024
#define Hh   16
#define DKk  64
#define Tt   1024
#define Vv   32000
#define Bb   2
#define DFF  4096
#define Mrows (Bb*Tt)   // 2048
#define MB(x) ((size_t)(x) << 20)

typedef __attribute__((ext_vector_type(16))) _Float16 v16h;
typedef __attribute__((ext_vector_type(8)))  _Float16 v8h;
typedef __attribute__((ext_vector_type(8)))  float    v8f;
typedef __attribute__((ext_vector_type(4)))  unsigned int u32x4;
typedef __attribute__((ext_vector_type(8)))  int          i32x8;
typedef __attribute__((ext_vector_type(4)))  int          i32x4;

static __device__ __forceinline__ v8f wmma_f16(const v16h& a, const v16h& b, const v8f& c) {
    return __builtin_amdgcn_wmma_f32_16x16x32_f16(false, a, false, b, (short)0, c, false, false);
}

// ---- Tensor Data Mover: 2-D f16 tile (global -> LDS), ISA cdna5 §8.3/8.4 ----
// Copies tile_rows rows of tile_k halfwords (row stride row_stride_elems halves)
// starting at gaddr into LDS at byte offset lds_off, packed densely.
static __device__ __forceinline__ void tdm_load_f16(unsigned lds_off, const void* gaddr,
                                                    unsigned tile_k, unsigned tile_rows,
                                                    unsigned row_stride_elems) {
    const unsigned long long ga = (unsigned long long)(uintptr_t)gaddr;
    u32x4 g0;
    g0.x = 1u;                                              // count=1, user mode
    g0.y = lds_off;                                         // lds_addr [63:32]
    g0.z = (unsigned)(ga & 0xffffffffull);                  // global_addr [95:64]
    g0.w = (unsigned)((ga >> 32) & 0x1ffffffull) | (2u << 30);  // addr[56:32] | type=2
    i32x8 g1;
    g1[0] = (int)(1u << 16);                                // wg_mask=0, data_size=1 (2B)
    g1[1] = (int)((row_stride_elems & 0xffffu) << 16);      // tensor_dim0[15:0] @bit48
    g1[2] = (int)(((row_stride_elems >> 16) & 0xffffu)      // tensor_dim0[31:16]
                  | ((tile_rows & 0xffffu) << 16));         // tensor_dim1[15:0] @bit80
    g1[3] = (int)(((tile_rows >> 16) & 0xffffu)             // tensor_dim1[31:16]
                  | ((tile_k & 0xffffu) << 16));            // tile_dim0 @bit112
    g1[4] = (int)(tile_rows & 0xffffu);                     // tile_dim1 @bit128, tile_dim2=0
    g1[5] = (int)row_stride_elems;                          // tensor_dim0_stride[31:0] @bit160
    g1[6] = 0;                                              // stride hi, tensor_dim1_stride=0
    g1[7] = 0;
    const i32x4 z4 = {0, 0, 0, 0};
    const i32x8 z8 = {0, 0, 0, 0, 0, 0, 0, 0};
    __builtin_amdgcn_tensor_load_to_lds(g0, g1, z4, z4, z8, 0);
}

// ---------------- weight pre-pack: fp32 [K][N] -> f16 tiled [ntile][kt][nn*32+kk] ----
// Tile = 32(K) x 128(N); packed so each GEMM B tile is one contiguous 8 KB blob
// already in WMMA fragment order ([n][k]).
__global__ __launch_bounds__(256) void pack_w(const float* __restrict__ W,
                                              _Float16* __restrict__ Wp,
                                              int K, int N) {
    const size_t p = (size_t)blockIdx.x * 256 + threadIdx.x;   // packed index
    const int Ktiles = K >> 5;
    const size_t tIdx = p >> 12;                // 4096 elements per tile
    const int within = (int)(p & 4095);
    const int nn = within >> 5, kk = within & 31;
    const int kt    = (int)(tIdx % Ktiles);
    const int ntile = (int)(tIdx / Ktiles);
    Wp[p] = (_Float16)W[(size_t)(kt * 32 + kk) * N + (size_t)ntile * 128 + nn];
}

// ---------------- embedding ----------------
__global__ __launch_bounds__(256) void embed_kernel(const int* __restrict__ x,
                                                    const float* __restrict__ ie,
                                                    const float* __restrict__ pe,
                                                    float* __restrict__ h) {
    int row = blockIdx.x;            // b*T + t
    int t   = row % Tt;
    int tok = x[row];
    for (int d = threadIdx.x; d < Dm; d += 256)
        h[(size_t)row * Dm + d] = ie[(size_t)tok * Dm + d] + pe[(size_t)t * Dm + d];
}

// ---------------- layernorm (fp32 in -> f16 out) ----------------
__global__ __launch_bounds__(256) void layernorm_kernel(const float* __restrict__ x,
                                                        const float* __restrict__ gamma,
                                                        const float* __restrict__ beta,
                                                        _Float16* __restrict__ out) {
    __shared__ float red[256];
    const int row = blockIdx.x;
    const int tid = threadIdx.x;
    const float* xr = x + (size_t)row * Dm;

    float v4[4];
    float s = 0.f;
#pragma unroll
    for (int i = 0; i < 4; ++i) { v4[i] = xr[tid + i * 256]; s += v4[i]; }
    red[tid] = s; __syncthreads();
    for (int st = 128; st > 0; st >>= 1) { if (tid < st) red[tid] += red[tid + st]; __syncthreads(); }
    const float mu = red[0] * (1.0f / Dm);
    __syncthreads();

    float ss = 0.f;
#pragma unroll
    for (int i = 0; i < 4; ++i) { float d = v4[i] - mu; ss += d * d; }
    red[tid] = ss; __syncthreads();
    for (int st = 128; st > 0; st >>= 1) { if (tid < st) red[tid] += red[tid + st]; __syncthreads(); }
    const float rs = rsqrtf(red[0] * (1.0f / Dm) + 1e-5f);

#pragma unroll
    for (int i = 0; i < 4; ++i) {
        int d = tid + i * 256;
        out[(size_t)row * Dm + d] = (_Float16)((v4[i] - mu) * rs * gamma[d] + beta[d]);
    }
}

// ---------------- generic WMMA GEMM (TDM-fed) ----------------
// C[M,N] = act(A_f16[M,K] @ Bp + bias) (+ residual); Bp = pre-packed f16 tiles.
// 256 thr = 8 waves as 2(M) x 4(N); wave tile 32x32 (4 WMMAs / K32 step);
// block tile 64(M) x 128(N). A and B tiles pulled into LDS by the Tensor Data
// Mover (wave 0 issues, TENSORcnt drained before the publishing barrier);
// double-buffered, 1 barrier per K step.
template <bool OUT_F16, bool GELU, bool RESID, bool HAS_BIAS>
__global__ __launch_bounds__(256) void gemm_wmma(const _Float16* __restrict__ A,
                                                 const _Float16* __restrict__ Bp,
                                                 const float* __restrict__ bias,
                                                 const float* __restrict__ resid,
                                                 void* __restrict__ out,
                                                 int K, int N) {
    __shared__ _Float16 Bs[2][128 * 32];      // [buf][n][k] f16, 8 KB each
    __shared__ _Float16 As[2][64 * 32];       // [buf][m][k] f16, 4 KB each
    const int tid  = threadIdx.x;
    const int wave = tid >> 5;                // 0..7
    const int lane = tid & 31;
    const int half = lane >> 4;
    const int ln   = lane & 15;
    const int kb   = half * 8;                // per-lane K offset in A/B fragments
    const int wm   = wave >> 2;               // 0..1  (M group)
    const int wn   = wave & 3;                // 0..3  (N group)
    const int mBlk = blockIdx.y * 64;
    const int nBlk = blockIdx.x * 128;
    const int Ktiles = K >> 5;
    const _Float16* BpBase = Bp + (size_t)blockIdx.x * Ktiles * 4096;

    const unsigned bs0 = (unsigned)(uintptr_t)(void*)&Bs[0][0];
    const unsigned bs1 = (unsigned)(uintptr_t)(void*)&Bs[1][0];
    const unsigned as0 = (unsigned)(uintptr_t)(void*)&As[0][0];
    const unsigned as1 = (unsigned)(uintptr_t)(void*)&As[1][0];

    v8f c[2][2];
#pragma unroll
    for (int mt = 0; mt < 2; ++mt)
#pragma unroll
        for (int nt = 0; nt < 2; ++nt)
#pragma unroll
            for (int r = 0; r < 8; ++r) c[mt][nt][r] = 0.f;

    // preload tile 0 into buffer 0 via TDM
    if (wave == 0) {
        tdm_load_f16(bs0, BpBase, 4096, 1, 4096);                        // B: 8 KB blob
        tdm_load_f16(as0, A + (size_t)mBlk * K, 32, 64, (unsigned)K);    // A: 64x32 tile
        __builtin_amdgcn_s_wait_tensorcnt(0);
    }
    __syncthreads();

    int cur = 0;
    for (int k0 = 0; k0 < K; k0 += 32) {
        // issue next-tile TDM copies into the other buffer (overlaps WMMA chain)
        if (wave == 0 && k0 + 32 < K) {
            tdm_load_f16(cur ? bs0 : bs1, BpBase + (size_t)((k0 >> 5) + 1) * 4096, 4096, 1, 4096);
            tdm_load_f16(cur ? as0 : as1, A + (size_t)mBlk * K + k0 + 32, 32, 64, (unsigned)K);
        }

        // A fragments from LDS: two 16-row M subtiles
        v16h a[2];
#pragma unroll
        for (int mt = 0; mt < 2; ++mt) {
            const int mrow = wm * 32 + mt * 16 + ln;
            const v8h a_lo = *(const v8h*)(As[cur] + mrow * 32 + kb);
            const v8h a_hi = *(const v8h*)(As[cur] + mrow * 32 + 16 + kb);
#pragma unroll
            for (int j = 0; j < 8; ++j) { a[mt][j] = a_lo[j]; a[mt][8 + j] = a_hi[j]; }
        }
        // B fragments from LDS: two 16-col N subtiles
        v16h b[2];
#pragma unroll
        for (int nt = 0; nt < 2; ++nt) {
            const int col = wn * 32 + nt * 16 + ln;
            const v8h b_lo = *(const v8h*)(Bs[cur] + col * 32 + kb);
            const v8h b_hi = *(const v8h*)(Bs[cur] + col * 32 + 16 + kb);
#pragma unroll
            for (int j = 0; j < 8; ++j) { b[nt][j] = b_lo[j]; b[nt][8 + j] = b_hi[j]; }
        }

#pragma unroll
        for (int mt = 0; mt < 2; ++mt)
#pragma unroll
            for (int nt = 0; nt < 2; ++nt)
                c[mt][nt] = wmma_f16(a[mt], b[nt], c[mt][nt]);

        if (wave == 0) __builtin_amdgcn_s_wait_tensorcnt(0);   // drain this iter's TDMs
        __syncthreads();                       // publishes buf[cur^1], retires buf[cur]
        cur ^= 1;
    }

    // epilogue: C row m = half*8 + r, col n = lane&15 within each 16x16 tile
#pragma unroll
    for (int mt = 0; mt < 2; ++mt)
#pragma unroll
        for (int nt = 0; nt < 2; ++nt)
#pragma unroll
            for (int r = 0; r < 8; ++r) {
                const int row = mBlk + wm * 32 + mt * 16 + half * 8 + r;
                const int col = nBlk + wn * 32 + nt * 16 + ln;
                float val = c[mt][nt][r];
                if constexpr (HAS_BIAS) val += bias[col];
                if constexpr (GELU)     val = 0.5f * val * (1.0f + erff(val * 0.70710678f));
                if constexpr (RESID)    val += resid[(size_t)row * N + col];
                if constexpr (OUT_F16)  ((_Float16*)out)[(size_t)row * N + col] = (_Float16)val;
                else                    ((float*)out)[(size_t)row * N + col] = val;
            }
}

// ---------------- causal flash attention ----------------
// one wave per (b, h, 16-query tile); q/k/v/y are f16 [B,T,H,DK] (== [B,T,D])
__global__ __launch_bounds__(32) void attn_kernel(const _Float16* __restrict__ q,
                                                  const _Float16* __restrict__ k,
                                                  const _Float16* __restrict__ v,
                                                  _Float16* __restrict__ y) {
    __shared__ _Float16 Pt[16 * 32];   // P tile, [row][kcol]
    __shared__ _Float16 Vt[64 * 32];   // V chunk transposed [dk][key]

    const int qt  = blockIdx.x;
    const int bh  = blockIdx.y;
    const int b   = bh / Hh;
    const int h   = bh % Hh;
    const int lane = threadIdx.x;
    const int half = lane >> 4;
    const int ln   = lane & 15;
    const int kb   = half * 8;

    // Q fragments (16 queries x 64 dk = two K32 A-fragments)
    const size_t qbase = ((size_t)(b * Tt + qt * 16 + ln)) * Dm + h * DKk;
    v16h aq0, aq1;
    {
        const v8h q0 = *(const v8h*)(q + qbase + kb);
        const v8h q1 = *(const v8h*)(q + qbase + 16 + kb);
        const v8h q2 = *(const v8h*)(q + qbase + 32 + kb);
        const v8h q3 = *(const v8h*)(q + qbase + 48 + kb);
#pragma unroll
        for (int j = 0; j < 8; ++j) { aq0[j] = q0[j]; aq0[8 + j] = q1[j];
                                      aq1[j] = q2[j]; aq1[8 + j] = q3[j]; }
    }

    float rm[8], rl[8];
    v8f o[4];
#pragma unroll
    for (int r = 0; r < 8; ++r) { rm[r] = -1e30f; rl[r] = 0.f; }
#pragma unroll
    for (int jv = 0; jv < 4; ++jv)
#pragma unroll
        for (int r = 0; r < 8; ++r) o[jv][r] = 0.f;

    const int qend = qt * 16 + 15;
    for (int c0 = 0; c0 <= qend; c0 += 32) {
        // stage V chunk (32 keys x 64 dk) transposed into LDS [dk][key]
        {
            const int key = c0 + lane;
            const v8h* vp = (const v8h*)(v + ((size_t)(b * Tt + key)) * Dm + h * DKk);
#pragma unroll
            for (int jj = 0; jj < 8; ++jj) {
                v8h vr = vp[jj];
#pragma unroll
                for (int e = 0; e < 8; ++e) Vt[(jj * 8 + e) * 32 + lane] = vr[e];
            }
        }
        __syncthreads();

        // S = Q K^T for two 16-key subtiles
        v8f s[2];
#pragma unroll
        for (int j = 0; j < 2; ++j) {
            const size_t kbase = ((size_t)(b * Tt + c0 + j * 16 + ln)) * Dm + h * DKk;
            v16h bk0, bk1;
            const v8h k0v = *(const v8h*)(k + kbase + kb);
            const v8h k1v = *(const v8h*)(k + kbase + 16 + kb);
            const v8h k2v = *(const v8h*)(k + kbase + 32 + kb);
            const v8h k3v = *(const v8h*)(k + kbase + 48 + kb);
#pragma unroll
            for (int e = 0; e < 8; ++e) { bk0[e] = k0v[e]; bk0[8 + e] = k1v[e];
                                          bk1[e] = k2v[e]; bk1[8 + e] = k3v[e]; }
            v8f t = {0.f, 0.f, 0.f, 0.f, 0.f, 0.f, 0.f, 0.f};
            t = wmma_f16(aq0, bk0, t);
            t = wmma_f16(aq1, bk1, t);
#pragma unroll
            for (int r = 0; r < 8; ++r) {
                const int qg = qt * 16 + half * 8 + r;
                const int kg = c0 + j * 16 + ln;
                float sv = t[r] * 0.125f;          // 1/sqrt(64)
                if (kg > qg) sv = -1e30f;
                t[r] = sv;
            }
            s[j] = t;
        }

        // online softmax: per-row stats live in (vgpr r, lane-half); reduce over 16 lanes
        float alpha[8];
        v8f p0, p1;
#pragma unroll
        for (int r = 0; r < 8; ++r) {
            float mx = fmaxf(s[0][r], s[1][r]);
#pragma unroll
            for (int off = 1; off <= 8; off <<= 1) mx = fmaxf(mx, __shfl_xor(mx, off, 32));
            const float nm = fmaxf(rm[r], mx);
            const float al = expf(rm[r] - nm);
            const float e0 = expf(s[0][r] - nm);
            const float e1 = expf(s[1][r] - nm);
            p0[r] = e0; p1[r] = e1;
            float rs = e0 + e1;
#pragma unroll
            for (int off = 1; off <= 8; off <<= 1) rs += __shfl_xor(rs, off, 32);
            rl[r] = rl[r] * al + rs;
            rm[r] = nm;
            alpha[r] = al;
        }
#pragma unroll
        for (int jv = 0; jv < 4; ++jv)
#pragma unroll
            for (int r = 0; r < 8; ++r) o[jv][r] *= alpha[r];

        // relayout P (C-layout -> A-fragment) through LDS
#pragma unroll
        for (int r = 0; r < 8; ++r) {
            const int row = half * 8 + r;
            Pt[row * 32 + ln]      = (_Float16)p0[r];
            Pt[row * 32 + 16 + ln] = (_Float16)p1[r];
        }
        __syncthreads();

        v16h aP;
        {
            const v8h p_lo = *(const v8h*)(Pt + ln * 32 + kb);
            const v8h p_hi = *(const v8h*)(Pt + ln * 32 + 16 + kb);
#pragma unroll
            for (int j = 0; j < 8; ++j) { aP[j] = p_lo[j]; aP[8 + j] = p_hi[j]; }
        }

        // O += P @ V  (4 dk tiles of 16)
#pragma unroll
        for (int jv = 0; jv < 4; ++jv) {
            const v8h b_lo = *(const v8h*)(Vt + (jv * 16 + ln) * 32 + kb);
            const v8h b_hi = *(const v8h*)(Vt + (jv * 16 + ln) * 32 + 16 + kb);
            v16h bV;
#pragma unroll
            for (int e = 0; e < 8; ++e) { bV[e] = b_lo[e]; bV[8 + e] = b_hi[e]; }
            o[jv] = wmma_f16(aP, bV, o[jv]);
        }
        __syncthreads();
    }

    // normalize + store y
#pragma unroll
    for (int jv = 0; jv < 4; ++jv)
#pragma unroll
        for (int r = 0; r < 8; ++r) {
            const int row = qt * 16 + half * 8 + r;
            const float val = o[jv][r] / rl[r];
            y[((size_t)(b * Tt + row)) * Dm + h * DKk + jv * 16 + ln] = (_Float16)val;
        }
}

// ---------------- host orchestration ----------------
extern "C" void kernel_launch(void* const* d_in, const int* in_sizes, int n_in,
                              void* d_out, int out_size, void* d_ws, size_t ws_size,
                              hipStream_t stream) {
    (void)in_sizes; (void)n_in; (void)out_size; (void)ws_size;
    const int*   x      = (const int*)  d_in[0];
    const float* ie     = (const float*)d_in[1];
    const float* pe     = (const float*)d_in[2];
    const float* wq     = (const float*)d_in[3];
    const float* bq     = (const float*)d_in[4];
    const float* wk     = (const float*)d_in[5];
    const float* bk     = (const float*)d_in[6];
    const float* wv     = (const float*)d_in[7];
    const float* bv     = (const float*)d_in[8];
    const float* wo     = (const float*)d_in[9];
    const float* bo     = (const float*)d_in[10];
    const float* w1     = (const float*)d_in[11];
    const float* b1     = (const float*)d_in[12];
    const float* w2     = (const float*)d_in[13];
    const float* b2     = (const float*)d_in[14];
    const float* ln1_s  = (const float*)d_in[15];
    const float* ln1_b  = (const float*)d_in[16];
    const float* ln2_s  = (const float*)d_in[17];
    const float* ln2_b  = (const float*)d_in[18];
    const float* lno_s  = (const float*)d_in[19];
    const float* lno_b  = (const float*)d_in[20];
    const float* w_out  = (const float*)d_in[21];

    char* ws = (char*)d_ws;
    float*     h    = (float*)ws;                       // 8 MB
    _Float16*  xn   = (_Float16*)(ws + MB(8));          // 4 MB
    _Float16*  qb   = (_Float16*)(ws + MB(12));         // 4 MB
    _Float16*  kbuf = (_Float16*)(ws + MB(16));         // 4 MB
    _Float16*  vb   = (_Float16*)(ws + MB(20));         // 4 MB
    _Float16*  yb   = (_Float16*)(ws + MB(24));         // 4 MB
    _Float16*  ff   = (_Float16*)(ws + MB(28));         // 16 MB
    // packed f16 weights (fragment-tile order)
    _Float16*  wqP  = (_Float16*)(ws + MB(44));         // 16 MB (8 x D*D)
    _Float16*  wkP  = (_Float16*)(ws + MB(60));
    _Float16*  wvP  = (_Float16*)(ws + MB(76));
    _Float16*  woP  = (_Float16*)(ws + MB(92));
    _Float16*  w1P  = (_Float16*)(ws + MB(108));        // 64 MB (8 x D*DFF)
    _Float16*  w2P  = (_Float16*)(ws + MB(172));        // 64 MB
    _Float16*  woutP= (_Float16*)(ws + MB(236));        // 62.5 MB

    const dim3 blkG(256), blkL(256), blkA(32);
    const dim3 grdD(Dm / 128, Mrows / 64);    // N=1024 GEMMs
    const dim3 grdF(DFF / 128, Mrows / 64);   // N=4096 GEMM
    const dim3 grdV(Vv / 128, Mrows / 64);    // logits GEMM
    const dim3 grdA(Tt / 16, Bb * Hh);        // attention
    const dim3 grdR(Mrows);
    const int  gDD = Dm * Dm / 256, gDF = Dm * DFF / 256, gDV = Dm * Vv / 256;

    // one-time weight convert + swizzle
    for (int i = 0; i < NL; ++i) {
        const size_t o1 = (size_t)i * Dm * Dm, o2 = (size_t)i * Dm * DFF;
        pack_w<<<gDD, blkG, 0, stream>>>(wq + o1, wqP + o1, Dm, Dm);
        pack_w<<<gDD, blkG, 0, stream>>>(wk + o1, wkP + o1, Dm, Dm);
        pack_w<<<gDD, blkG, 0, stream>>>(wv + o1, wvP + o1, Dm, Dm);
        pack_w<<<gDD, blkG, 0, stream>>>(wo + o1, woP + o1, Dm, Dm);
        pack_w<<<gDF, blkG, 0, stream>>>(w1 + o2, w1P + o2, Dm, DFF);
        pack_w<<<gDF, blkG, 0, stream>>>(w2 + o2, w2P + o2, DFF, Dm);
    }
    pack_w<<<gDV, blkG, 0, stream>>>(w_out, woutP, Dm, Vv);

    embed_kernel<<<grdR, blkL, 0, stream>>>(x, ie, pe, h);

    for (int i = 0; i < NL; ++i) {
        const size_t wOff = (size_t)i * Dm * Dm;
        const size_t fOff = (size_t)i * Dm * DFF;
        const size_t bOff = (size_t)i * Dm;
        // attention block
        layernorm_kernel<<<grdR, blkL, 0, stream>>>(h, ln1_s + bOff, ln1_b + bOff, xn);
        gemm_wmma<true, false, false, true><<<grdD, blkG, 0, stream>>>(xn, wqP + wOff, bq + bOff, nullptr, qb, Dm, Dm);
        gemm_wmma<true, false, false, true><<<grdD, blkG, 0, stream>>>(xn, wkP + wOff, bk + bOff, nullptr, kbuf, Dm, Dm);
        gemm_wmma<true, false, false, true><<<grdD, blkG, 0, stream>>>(xn, wvP + wOff, bv + bOff, nullptr, vb, Dm, Dm);
        attn_kernel<<<grdA, blkA, 0, stream>>>(qb, kbuf, vb, yb);
        gemm_wmma<false, false, true, true><<<grdD, blkG, 0, stream>>>(yb, woP + wOff, bo + bOff, h, h, Dm, Dm);
        // FFN block
        layernorm_kernel<<<grdR, blkL, 0, stream>>>(h, ln2_s + bOff, ln2_b + bOff, xn);
        gemm_wmma<true, true, false, true><<<grdF, blkG, 0, stream>>>(xn, w1P + fOff, b1 + (size_t)i * DFF, nullptr, ff, Dm, DFF);
        gemm_wmma<false, false, true, true><<<grdD, blkG, 0, stream>>>(ff, w2P + fOff, b2 + bOff, h, h, DFF, Dm);
    }

    layernorm_kernel<<<grdR, blkL, 0, stream>>>(h, lno_s, lno_b, xn);
    gemm_wmma<false, false, false, false><<<grdV, blkG, 0, stream>>>(xn, woutP, nullptr, nullptr, (float*)d_out, Dm, Vv);
}
